// RNNModule_50027779064124
// MI455X (gfx1250) — matrix operand order repaired and checked by reading, not compiled, not run on hardware
//
#include <hip/hip_runtime.h>

typedef __attribute__((ext_vector_type(16))) __bf16 v16bf;
typedef __attribute__((ext_vector_type(8)))  float  v8f;
typedef __attribute__((ext_vector_type(4)))  float  f32x4;

#define B_  128
#define T_  2048
#define NI_ 128
#define NH_ 128

// ---- fragment loaders -------------------------------------------------------
// A-matrix 16x32 bf16 layout (wave32): lanes 0-15 = rows, lane holds
// K {base..base+7} and {base+16..base+23} with base = 0 (lo half) or 8 (hi).
// Non-temporal: inputs are single-use; keep L2 free for igates.
__device__ __forceinline__ v16bf load_a_f32_nt(const float* __restrict__ ap) {
  const f32x4 p0 = __builtin_nontemporal_load((const f32x4*)(ap + 0));
  const f32x4 p1 = __builtin_nontemporal_load((const f32x4*)(ap + 4));
  const f32x4 p2 = __builtin_nontemporal_load((const f32x4*)(ap + 16));
  const f32x4 p3 = __builtin_nontemporal_load((const f32x4*)(ap + 20));
  v16bf a;
#pragma unroll
  for (int i = 0; i < 4; ++i) {
    a[i]      = (__bf16)p0[i];
    a[4 + i]  = (__bf16)p1[i];
    a[8 + i]  = (__bf16)p2[i];
    a[12 + i] = (__bf16)p3[i];
  }
  return a;
}
__device__ __forceinline__ v16bf load_a_bf16(const __bf16* ap) {
  v16bf a;
#pragma unroll
  for (int i = 0; i < 8; ++i) a[i] = ap[i];
#pragma unroll
  for (int i = 0; i < 8; ++i) a[8 + i] = ap[16 + i];
  return a;
}
// B-matrix 32x16 bf16 layout: lane holds K {base..base+7} and
// {base+16..base+23} of column N = lane&15, base = 0 (lo half) or 8 (hi).
// B[k][n] = W[n][k] -> runs of row n of W (hot in L0/L2, regular temporal).
__device__ __forceinline__ v16bf load_b_f32(const float* __restrict__ bp) {
  v16bf b;
#pragma unroll
  for (int i = 0; i < 16; ++i) b[i] = (__bf16)bp[(i < 8) ? i : (i + 8)];
  return b;
}

__device__ __forceinline__ float fast_sigmoid(float x) {
  return __builtin_amdgcn_rcpf(1.0f + __expf(-x));
}

// ---- phase 1: igates = inputs * W_in^T + (b_in + b_hh) ----------------------
// grid: (B*T)/128 blocks, 256 threads (8 waves); wave = 16-row strip x 128 cols
__global__ __launch_bounds__(256) void igates_kernel(
    const float* __restrict__ inputs, const float* __restrict__ W_in,
    const float* __restrict__ b_in, const float* __restrict__ b_hh,
    float* __restrict__ out) {
  const int  lane = threadIdx.x & 31;
  const int  wave = threadIdx.x >> 5;
  const bool lo   = lane < 16;
  const int  m    = lane & 15;
  const size_t r0 = (size_t)blockIdx.x * 128 + (size_t)wave * 16;

  v8f c[8];
#pragma unroll
  for (int nt = 0; nt < 8; ++nt) {
    const int   n    = nt * 16 + m;         // this lane's output column
    const float bias = b_in[n] + b_hh[n];   // same for all 8 row-regs
#pragma unroll
    for (int j = 0; j < 8; ++j) c[nt][j] = bias;
  }

  const float* arow = inputs + (r0 + (size_t)m) * NI_;
#pragma unroll
  for (int kc = 0; kc < 4; ++kc) {
    v16bf a = load_a_f32_nt(arow + kc * 32 + (lo ? 0 : 8));
#pragma unroll
    for (int nt = 0; nt < 8; ++nt) {
      const float* bp =
          W_in + (size_t)(nt * 16 + m) * NI_ + kc * 32 + (lo ? 0 : 8);
      v16bf b = load_b_f32(bp);
      c[nt] = __builtin_amdgcn_wmma_f32_16x16x32_bf16(
          false, a, false, b, (short)0, c[nt], false, false);
    }
  }

  // C layout: VGPR j -> row M = j (+8 for hi lanes), column = lane&15.
  // Regular-temporal stores: igates must stay L2-resident for the scan.
  const size_t rbase = r0 + (lo ? 0 : 8);
#pragma unroll
  for (int nt = 0; nt < 8; ++nt) {
    const int n = nt * 16 + m;
#pragma unroll
    for (int j = 0; j < 8; ++j) out[(rbase + j) * NH_ + n] = c[nt][j];
  }
}

// ---- phase 2: sequential scan, h = sigmoid(ig_t + h * W_hh^T) ---------------
// grid: B/16 = 8 blocks, 256 threads (8 waves). Block owns 16 batch rows;
// wave owns 16 output columns with resident W_hh^T fragments. h is shared
// across waves via a double-buffered bf16 LDS tile (stride 136 -> no bank
// conflicts). One barrier per timestep; igates(t+1) prefetched during t.
__global__ __launch_bounds__(256) void scan_kernel(
    const float* __restrict__ hidden, const float* __restrict__ W_hh,
    float* __restrict__ io /* igates in, h out, in place */) {
  __shared__ __bf16 hbuf[2][16][136];

  const int  tid  = threadIdx.x;
  const int  lane = tid & 31;
  const int  wave = tid >> 5;
  const bool lo   = lane < 16;
  const int  m    = lane & 15;
  const int  b0   = blockIdx.x * 16;
  const int  n    = wave * 16 + m;  // this lane's output column

  // stage h(0) = hidden into buffer 0
  for (int idx = tid; idx < 16 * 128; idx += 256) {
    const int r = idx >> 7, cc = idx & 127;
    hbuf[0][r][cc] = (__bf16)hidden[(size_t)(b0 + r) * NH_ + cc];
  }

  // resident B fragments of W_hh^T for this wave's column slab
  v16bf wf[4];
#pragma unroll
  for (int kc = 0; kc < 4; ++kc)
    wf[kc] = load_b_f32(W_hh + (size_t)n * NH_ + kc * 32 + (lo ? 0 : 8));

  const int mr = lo ? 0 : 8;  // C row offset for this half-wave
  size_t rowoff[8];           // io index of (b0+mr+j, t=0, n)
#pragma unroll
  for (int j = 0; j < 8; ++j)
    rowoff[j] = (size_t)(b0 + mr + j) * T_ * NH_ + n;

  __syncthreads();

  float ig[8];
#pragma unroll
  for (int j = 0; j < 8; ++j) ig[j] = io[rowoff[j]];  // prefetch t = 0

#pragma unroll 1
  for (int t = 0; t < T_; ++t) {
    v8f c;
#pragma unroll
    for (int j = 0; j < 8; ++j) c[j] = ig[j];

    // branchless prefetch of igates(t+1) behind the WMMAs (clamped at end;
    // the redundant last-step load precedes this wave's store in order)
    const size_t tn = (t < T_ - 1) ? (size_t)(t + 1) : (size_t)(T_ - 1);
#pragma unroll
    for (int j = 0; j < 8; ++j) ig[j] = io[rowoff[j] + tn * NH_];

    const __bf16* hp = &hbuf[t & 1][m][0];
#pragma unroll
    for (int kc = 0; kc < 4; ++kc) {
      v16bf a = load_a_bf16(hp + kc * 32 + (lo ? 0 : 8));
      c = __builtin_amdgcn_wmma_f32_16x16x32_bf16(
          false, a, false, wf[kc], (short)0, c, false, false);
    }

#pragma unroll
    for (int j = 0; j < 8; ++j) {
      const float h = fast_sigmoid(c[j]);
      io[rowoff[j] + (size_t)t * NH_] = h;          // overwrite igates with h
      hbuf[(t + 1) & 1][mr + j][n] = (__bf16)h;     // publish for next step
    }
    __syncthreads();
  }
}

// ---- host launcher ----------------------------------------------------------
extern "C" void kernel_launch(void* const* d_in, const int* in_sizes, int n_in,
                              void* d_out, int out_size, void* d_ws,
                              size_t ws_size, hipStream_t stream) {
  const float* inputs = (const float*)d_in[0];
  const float* hidden = (const float*)d_in[1];
  const float* W_in   = (const float*)d_in[2];
  const float* b_in   = (const float*)d_in[3];
  const float* W_hh   = (const float*)d_in[4];
  const float* b_hh   = (const float*)d_in[5];
  float* out = (float*)d_out;

  (void)in_sizes; (void)n_in; (void)out_size; (void)d_ws; (void)ws_size;

  igates_kernel<<<(B_ * T_) / 128, 256, 0, stream>>>(inputs, W_in, b_in, b_hh,
                                                     out);
  scan_kernel<<<B_ / 16, 256, 0, stream>>>(hidden, W_hh, out);
}
